// TransformerBlock_197568496226
// MI455X (gfx1250) — matrix operand order repaired and checked by reading, not compiled, not run on hardware
//
#include <hip/hip_runtime.h>
#include <hip/hip_bf16.h>

// ---------------------------------------------------------------------------
// Transformer block forward for MI455X (gfx1250).
// wave32 + v_wmma_f32_16x16x32_f16 everywhere; GEMMs use the CDNA5 async
// global->LDS engine (global_load_async_to_lds_b128 / ASYNCcnt) with double
// buffering; attention is a flash-style online-softmax over WMMA tiles.
// ---------------------------------------------------------------------------

typedef __attribute__((ext_vector_type(16))) _Float16 v16h;
typedef __attribute__((ext_vector_type(8)))  _Float16 v8h;
typedef __attribute__((ext_vector_type(8)))  float    v8f;

#define EMBED   1024
#define HEADS   16
#define HEAD_D  64
#define FFN     4096
#define SEQ     2048
#define BATCH   4
#define TOKENS  (BATCH * SEQ)   // 8192

// -------------------------- WMMA helpers -----------------------------------
__device__ __forceinline__ v8f wmma_f16(v16h a, v16h b, v8f c) {
  return __builtin_amdgcn_wmma_f32_16x16x32_f16(
      /*neg_a=*/false, a, /*neg_b=*/false, b,
      /*c_mod=*/(short)0, c, /*reuse_a=*/false, /*reuse_b=*/false);
}

// A fragment, tile [M=16][K=32] row-major, ld in halves.
// ISA 16-bit A layout: lanes 0-15 (M=lane) hold K=0..7 & 16..23,
// lanes 16-31 hold K=8..15 & 24..31 -> two contiguous 16-byte runs.
__device__ __forceinline__ v16h frag_a_ld(const _Float16* t, int ld) {
  const int lane = threadIdx.x & 31;
  const int m  = lane & 15;
  const int kb = (lane < 16) ? 0 : 8;
  const _Float16* p = t + m * ld + kb;
  v8h lo = *(const v8h*)(p);
  v8h hi = *(const v8h*)(p + 16);
  return __builtin_shufflevector(lo, hi, 0,1,2,3,4,5,6,7,8,9,10,11,12,13,14,15);
}

// B fragment from a TRANSPOSED tile [N=16][K=32] row-major, ld in halves.
// 16-bit B layout matching A: lanes 0-15 (N=lane) hold K=0..7 & 16..23,
// lanes 16-31 hold K=8..15 & 24..31 -> two contiguous 16-byte runs.
__device__ __forceinline__ v16h frag_bT_ld(const _Float16* t, int ld) {
  const int lane = threadIdx.x & 31;
  const int n  = lane & 15;
  const int kb = (lane < 16) ? 0 : 8;
  const _Float16* p = t + n * ld + kb;
  v8h lo = *(const v8h*)(p);
  v8h hi = *(const v8h*)(p + 16);
  return __builtin_shufflevector(lo, hi, 0,1,2,3,4,5,6,7,8,9,10,11,12,13,14,15);
}
// C/D layout: element r (0..7) -> row M = r + 8*(lane>=16), col N = lane&15.

// -------------------------- async global -> LDS ----------------------------
// CDNA5 async copy: 16B per lane, tracked by ASYNCcnt (no VGPR data path).
__device__ __forceinline__ void async_b128(unsigned lds_off, const void* gaddr) {
  asm volatile("global_load_async_to_lds_b128 %0, %1, off"
               :: "v"(lds_off), "v"(gaddr) : "memory");
}
__device__ __forceinline__ void wait_async0() {
  asm volatile("s_wait_asynccnt 0x0" ::: "memory");
}

// -------------------------- weight conversion ------------------------------
// f32 [R][C] -> f16 transposed [C][R], LDS-tiled so both sides are coalesced.
__global__ void cvt_t_kernel(const float* __restrict__ s,
                             _Float16* __restrict__ d, int R, int C) {
  __shared__ _Float16 t[32][33];
  const int rb = blockIdx.y * 32, cb = blockIdx.x * 32;
  const int tx = threadIdx.x, ty = threadIdx.y;      // (32, 8)
#pragma unroll
  for (int j = 0; j < 32; j += 8)
    t[ty + j][tx] = (_Float16)s[(size_t)(rb + ty + j) * C + cb + tx];
  __syncthreads();
#pragma unroll
  for (int j = 0; j < 32; j += 8)
    d[(size_t)(cb + ty + j) * R + rb + tx] = t[tx][ty + j];
}

// 64x64 f32 -> f16 transposed (per-head projection weights)
__global__ void cvt_t64_kernel(const float* __restrict__ s,
                               _Float16* __restrict__ d) {
  int i = blockIdx.x * blockDim.x + threadIdx.x;     // 4096 threads
  int r = i >> 6, c = i & 63;
  d[c * 64 + r] = (_Float16)s[r * 64 + c];
}

// -------------------------- LayerNorm --------------------------------------
__global__ void ln_kernel(const float* __restrict__ in,
                          const float* __restrict__ gam,
                          const float* __restrict__ bet,
                          _Float16* __restrict__ out) {
  const int row = blockIdx.x;
  const float* p = in + (size_t)row * EMBED;
  float vals[4];
  float s = 0.f, s2 = 0.f;
#pragma unroll
  for (int i = 0; i < 4; ++i) {
    float v = p[threadIdx.x + i * 256];
    vals[i] = v; s += v; s2 += v * v;
  }
#pragma unroll
  for (int off = 16; off > 0; off >>= 1) {
    s  += __shfl_down(s,  off);
    s2 += __shfl_down(s2, off);
  }
  __shared__ float r1[8], r2[8];
  const int lane = threadIdx.x & 31, wid = threadIdx.x >> 5;
  if (lane == 0) { r1[wid] = s; r2[wid] = s2; }
  __syncthreads();
  if (threadIdx.x == 0) {
    float a = 0.f, b = 0.f;
#pragma unroll
    for (int w = 0; w < 8; ++w) { a += r1[w]; b += r2[w]; }
    r1[0] = a; r2[0] = b;
  }
  __syncthreads();
  const float mean = r1[0] * (1.0f / EMBED);
  const float var  = r2[0] * (1.0f / EMBED) - mean * mean;
  const float rs   = rsqrtf(var + 1e-5f);
#pragma unroll
  for (int i = 0; i < 4; ++i) {
    int c = threadIdx.x + i * 256;
    out[(size_t)row * EMBED + c] =
        (_Float16)((vals[i] - mean) * rs * gam[c] + bet[c]);
  }
}

// -------------------------- QKV projection ---------------------------------
// One wave per (16-token tile, head).  q scaled by 1/sqrt(EMBED);
// v stored transposed per (n,h): vT[((n*H+h)*64 + d) * SEQ + s].
__global__ void qkv_kernel(const _Float16* __restrict__ xn,
                           const _Float16* __restrict__ WqT,
                           const _Float16* __restrict__ WkT,
                           const _Float16* __restrict__ WvT,
                           _Float16* __restrict__ q,
                           _Float16* __restrict__ k,
                           _Float16* __restrict__ vT) {
  const int bid = blockIdx.x;
  const int h   = bid & (HEADS - 1);
  const int t0  = (bid / HEADS) * 16;
  const int lane = threadIdx.x;
  const int n    = t0 >> 11;
  const int s_in = t0 - (n << 11);
  const int mb   = (lane >> 4) * 8;
  const int c0   = lane & 15;

  const _Float16* ap = xn + (size_t)t0 * EMBED + h * HEAD_D;
  v16h a0 = frag_a_ld(ap, EMBED);
  v16h a1 = frag_a_ld(ap + 32, EMBED);

#pragma unroll
  for (int nc = 0; nc < 4; ++nc) {
    const int nco = nc * 16;
    v8f cq, ck, cv;
#pragma unroll
    for (int r = 0; r < 8; ++r) { cq[r] = 0.f; ck[r] = 0.f; cv[r] = 0.f; }
    cq = wmma_f16(a0, frag_bT_ld(WqT + nco * 64,      64), cq);
    cq = wmma_f16(a1, frag_bT_ld(WqT + nco * 64 + 32, 64), cq);
    ck = wmma_f16(a0, frag_bT_ld(WkT + nco * 64,      64), ck);
    ck = wmma_f16(a1, frag_bT_ld(WkT + nco * 64 + 32, 64), ck);
    cv = wmma_f16(a0, frag_bT_ld(WvT + nco * 64,      64), cv);
    cv = wmma_f16(a1, frag_bT_ld(WvT + nco * 64 + 32, 64), cv);
#pragma unroll
    for (int r = 0; r < 8; ++r) {
      const size_t row = (size_t)(t0 + mb + r);
      q[row * EMBED + h * HEAD_D + nco + c0] = (_Float16)(cq[r] * 0.03125f);
      k[row * EMBED + h * HEAD_D + nco + c0] = (_Float16)ck[r];
      vT[((size_t)(n * HEADS + h) * HEAD_D + nco + c0) * SEQ + (s_in + mb + r)] =
          (_Float16)cv[r];
    }
  }
}

// -------------------------- flash attention --------------------------------
__global__ void flash_kernel(const _Float16* __restrict__ q,
                             const _Float16* __restrict__ k,
                             const _Float16* __restrict__ vT,
                             _Float16* __restrict__ ctx) {
  const int bid = blockIdx.x;
  const int h   = bid & (HEADS - 1);
  const int t0  = (bid / HEADS) * 16;
  const int n   = t0 >> 11;
  const int s0  = n << 11;
  const int lane = threadIdx.x;
  const int mb   = (lane >> 4) * 8;
  const int c0   = lane & 15;

  __shared__ _Float16 lP[16 * 32];           // P: C-layout -> A-layout bounce

  const _Float16* qp = q + (size_t)t0 * EMBED + h * HEAD_D;
  const v16h aq0 = frag_a_ld(qp,      EMBED);
  const v16h aq1 = frag_a_ld(qp + 32, EMBED);

  const _Float16* kp  = k  + (size_t)s0 * EMBED + h * HEAD_D;
  const _Float16* vtb = vT + (size_t)(n * HEADS + h) * HEAD_D * SEQ;

  v8f acc[4];
  float m8[8], l8[8];
#pragma unroll
  for (int r = 0; r < 8; ++r) { m8[r] = -1e30f; l8[r] = 0.f; }
#pragma unroll
  for (int nc = 0; nc < 4; ++nc)
#pragma unroll
    for (int r = 0; r < 8; ++r) acc[nc][r] = 0.f;

  for (int kt = 0; kt < SEQ; kt += 32) {
    v8f sc0, sc1;
#pragma unroll
    for (int r = 0; r < 8; ++r) { sc0[r] = 0.f; sc1[r] = 0.f; }
    const _Float16* k0 = kp + (size_t)kt * EMBED;
    const _Float16* k1 = kp + (size_t)(kt + 16) * EMBED;
    sc0 = wmma_f16(aq0, frag_bT_ld(k0,      EMBED), sc0);
    sc0 = wmma_f16(aq1, frag_bT_ld(k0 + 32, EMBED), sc0);
    sc1 = wmma_f16(aq0, frag_bT_ld(k1,      EMBED), sc1);
    sc1 = wmma_f16(aq1, frag_bT_ld(k1 + 32, EMBED), sc1);
    if (kt + 32 < SEQ)
      __builtin_prefetch(kp + (size_t)(kt + 32) * EMBED, 0, 0);

    float scl[8];
#pragma unroll
    for (int r = 0; r < 8; ++r) {
      float t = fmaxf(sc0[r], sc1[r]);
#pragma unroll
      for (int off = 1; off < 16; off <<= 1) t = fmaxf(t, __shfl_xor(t, off));
      const float mn = fmaxf(m8[r], t);
      const float s  = __expf(m8[r] - mn);
      const float p0 = __expf(sc0[r] - mn);
      const float p1 = __expf(sc1[r] - mn);
      sc0[r] = p0; sc1[r] = p1;
      float rs = p0 + p1;
#pragma unroll
      for (int off = 1; off < 16; off <<= 1) rs += __shfl_xor(rs, off);
      l8[r] = l8[r] * s + rs;
      m8[r] = mn;
      scl[r] = s;
    }
#pragma unroll
    for (int nc = 0; nc < 4; ++nc)
#pragma unroll
      for (int r = 0; r < 8; ++r) acc[nc][r] *= scl[r];

#pragma unroll
    for (int r = 0; r < 8; ++r) {
      lP[(mb + r) * 32 + c0]      = (_Float16)sc0[r];
      lP[(mb + r) * 32 + 16 + c0] = (_Float16)sc1[r];
    }
    __syncthreads();                         // single-wave WG -> s_nop
    const v16h pA = frag_a_ld(lP, 32);

#pragma unroll
    for (int nc = 0; nc < 4; ++nc)
      acc[nc] = wmma_f16(pA, frag_bT_ld(vtb + (size_t)(nc * 16) * SEQ + kt, SEQ),
                         acc[nc]);
    __syncthreads();
  }

#pragma unroll
  for (int nc = 0; nc < 4; ++nc)
#pragma unroll
    for (int r = 0; r < 8; ++r) {
      const size_t row = (size_t)(t0 + mb + r);
      ctx[row * EMBED + h * HEAD_D + nc * 16 + c0] =
          (_Float16)(acc[nc][r] / l8[r]);
    }
}

// -------------------------- async double-buffered GEMM ---------------------
// C[M x N] = A[M x K] * BT[N x K]^T, B pre-transposed so A and B tiles have
// identical [64 rows x 64 bytes] shape, both staged by async b128 copies.
//   mode 0/2: out32 = acc + bias[col] + resid
//   mode 1  : out16 = gelu(acc + bias[col])
// Block: 128 threads (4 waves), 64x64 tile, double-buffered K steps of 32.
__global__ void gemm_kernel(const _Float16* __restrict__ A,
                            const _Float16* __restrict__ BT,
                            const float* __restrict__ bias,
                            const float* __restrict__ resid,
                            float* __restrict__ out32,
                            _Float16* __restrict__ out16,
                            int K, int N, int mode) {
  __shared__ _Float16 As[2][64 * 32];
  __shared__ _Float16 Bs[2][64 * 32];

  const int m0 = blockIdx.x * 64;
  const int n0 = blockIdx.y * 64;
  const int tid = threadIdx.x;
  const int lane = tid & 31, w = tid >> 5;

  // Each thread copies two 16B chunks of A and two of BT per tile.
  // chunk id c in [0,256): row = c>>2, 8-half segment = (c&3)*8.
  const int c0i = tid * 2, c1i = tid * 2 + 1;
  const int ar0 = c0i >> 2, ao0 = (c0i & 3) * 8;
  const int ar1 = c1i >> 2, ao1 = (c1i & 3) * 8;

  auto stage = [&](int kk, int b) {
    async_b128((unsigned)(size_t)&As[b][ar0 * 32 + ao0],
               A + (size_t)(m0 + ar0) * K + kk + ao0);
    async_b128((unsigned)(size_t)&As[b][ar1 * 32 + ao1],
               A + (size_t)(m0 + ar1) * K + kk + ao1);
    async_b128((unsigned)(size_t)&Bs[b][ar0 * 32 + ao0],
               BT + (size_t)(n0 + ar0) * K + kk + ao0);
    async_b128((unsigned)(size_t)&Bs[b][ar1 * 32 + ao1],
               BT + (size_t)(n0 + ar1) * K + kk + ao1);
  };

  v8f acc[4];
#pragma unroll
  for (int nc = 0; nc < 4; ++nc)
#pragma unroll
    for (int r = 0; r < 8; ++r) acc[nc][r] = 0.f;

  stage(0, 0);
  for (int kk = 0, it = 0; kk < K; kk += 32, ++it) {
    const int cur = it & 1;
    wait_async0();        // own tile landed in LDS
    __syncthreads();      // everyone's tile landed; prev buffer fully consumed
    if (kk + 32 < K) stage(kk + 32, cur ^ 1);   // prefetch next (other buffer)

    const v16h a = frag_a_ld(&As[cur][16 * w * 32], 32);
#pragma unroll
    for (int nc = 0; nc < 4; ++nc) {
      const v16h b = frag_bT_ld(&Bs[cur][(nc * 16) * 32], 32);
      acc[nc] = wmma_f16(a, b, acc[nc]);
    }
  }

  const int mb = (lane >> 4) * 8;
  const int cc = lane & 15;
#pragma unroll
  for (int nc = 0; nc < 4; ++nc)
#pragma unroll
    for (int r = 0; r < 8; ++r) {
      const size_t row = (size_t)(m0 + 16 * w + mb + r);
      const int    col = n0 + nc * 16 + cc;
      float v = acc[nc][r] + bias[col];
      if (mode == 1) {
        v = 0.5f * v * (1.0f + erff(v * 0.70710678118f));   // exact GELU
        out16[row * N + col] = (_Float16)v;
      } else {
        out32[row * N + col] = v + resid[row * N + col];
      }
    }
}

// -------------------------- host launch ------------------------------------
extern "C" void kernel_launch(void* const* d_in, const int* in_sizes, int n_in,
                              void* d_out, int out_size, void* d_ws, size_t ws_size,
                              hipStream_t stream) {
  const float* x     = (const float*)d_in[0];
  const float* Wq    = (const float*)d_in[1];
  const float* Wk    = (const float*)d_in[2];
  const float* Wv    = (const float*)d_in[3];
  const float* Wo    = (const float*)d_in[4];
  const float* bo    = (const float*)d_in[5];
  const float* ln1_g = (const float*)d_in[6];
  const float* ln1_b = (const float*)d_in[7];
  const float* ln2_g = (const float*)d_in[8];
  const float* ln2_b = (const float*)d_in[9];
  const float* W1    = (const float*)d_in[10];
  const float* b1    = (const float*)d_in[11];
  const float* W2    = (const float*)d_in[12];
  const float* b2    = (const float*)d_in[13];
  float* out = (float*)d_out;

  char* ws = (char*)d_ws;
  const size_t fp16_plane = (size_t)TOKENS * EMBED * sizeof(_Float16);  // 16 MB

  // Region A (64 MB): xn, q, k, vT ; later aliased by h1 (TOKENS x FFN f16).
  _Float16* xnh = (_Float16*)(ws);
  _Float16* qh  = (_Float16*)(ws + fp16_plane);
  _Float16* kh  = (_Float16*)(ws + 2 * fp16_plane);
  _Float16* vTh = (_Float16*)(ws + 3 * fp16_plane);
  _Float16* h1  = (_Float16*)(ws);                       // alias (64 MB)
  size_t off = 4 * fp16_plane;
  _Float16* ctxh = (_Float16*)(ws + off);                // 16 MB
  _Float16* anh  = ctxh;                                 // alias (ctx dead)
  off += fp16_plane;
  float* attn = (float*)(ws + off);                      // 32 MB f32
  off += (size_t)TOKENS * EMBED * sizeof(float);
  _Float16* WoT = (_Float16*)(ws + off); off += (size_t)EMBED * EMBED * 2;
  _Float16* W1T = (_Float16*)(ws + off); off += (size_t)EMBED * FFN * 2;
  _Float16* W2T = (_Float16*)(ws + off); off += (size_t)FFN * EMBED * 2;
  _Float16* WqT = (_Float16*)(ws + off); off += 64 * 64 * 2;
  _Float16* WkT = (_Float16*)(ws + off); off += 64 * 64 * 2;
  _Float16* WvT = (_Float16*)(ws + off); off += 64 * 64 * 2;
  (void)ws_size; (void)in_sizes; (void)n_in; (void)out_size;

  // 1) weight conversion (f32 -> f16, transposed to [N][K])
  cvt_t_kernel<<<dim3(EMBED / 32, EMBED / 32), dim3(32, 8), 0, stream>>>(
      Wo, WoT, EMBED, EMBED);
  cvt_t_kernel<<<dim3(FFN / 32, EMBED / 32), dim3(32, 8), 0, stream>>>(
      W1, W1T, EMBED, FFN);
  cvt_t_kernel<<<dim3(EMBED / 32, FFN / 32), dim3(32, 8), 0, stream>>>(
      W2, W2T, FFN, EMBED);
  cvt_t64_kernel<<<16, 256, 0, stream>>>(Wq, WqT);
  cvt_t64_kernel<<<16, 256, 0, stream>>>(Wk, WkT);
  cvt_t64_kernel<<<16, 256, 0, stream>>>(Wv, WvT);

  // 2) pre-LN
  ln_kernel<<<TOKENS, 256, 0, stream>>>(x, ln1_g, ln1_b, xnh);

  // 3) QKV projections
  qkv_kernel<<<(TOKENS / 16) * HEADS, 32, 0, stream>>>(xnh, WqT, WkT, WvT,
                                                       qh, kh, vTh);

  // 4) flash attention
  flash_kernel<<<(TOKENS / 16) * HEADS, 32, 0, stream>>>(qh, kh, vTh, ctxh);

  // 5) output projection + residual 1 -> attn (f32)
  gemm_kernel<<<dim3(TOKENS / 64, EMBED / 64), 128, 0, stream>>>(
      ctxh, WoT, bo, x, attn, nullptr, EMBED, EMBED, 0);

  // 6) LN2 -> anh (f16)
  ln_kernel<<<TOKENS, 256, 0, stream>>>(attn, ln2_g, ln2_b, anh);

  // 7) FFN up + GELU -> h1 (f16)
  gemm_kernel<<<dim3(TOKENS / 64, FFN / 64), 128, 0, stream>>>(
      anh, W1T, b1, nullptr, nullptr, h1, EMBED, FFN, 1);

  // 8) FFN down + residual 2 -> d_out (f32)
  gemm_kernel<<<dim3(TOKENS / 64, EMBED / 64), 128, 0, stream>>>(
      h1, W2T, b2, attn, out, nullptr, FFN, EMBED, 2);
}
